// MiniBatchOTLoss_62569083568791
// MI455X (gfx1250) — compile-verified, matrix operand order checked by reading, not run on hardware
//
#include <hip/hip_runtime.h>
#include <hip/hip_bf16.h>
#include <stdint.h>

// MI455X / gfx1250, wave32. Matrix math via V_WMMA_F32_16X16X32_BF16,
// tile staging via GLOBAL_LOAD_ASYNC_TO_LDS_B128 (ASYNCcnt) when available.

typedef __attribute__((ext_vector_type(16))) __bf16 v16bf;
typedef __attribute__((ext_vector_type(8)))  float  v8f;

#define SINK_ITERS 100
#define LP 40   // LDS pitch in bf16 elems (32 + 8 pad -> 80B rows, 16B-aligned, conflict-free)

#if defined(__AMDGCN__) && __has_builtin(__builtin_amdgcn_global_load_async_to_lds_b128) && \
    __has_builtin(__builtin_amdgcn_s_wait_asynccnt)
#define USE_ASYNC 1
#else
#define USE_ASYNC 0
#endif

#if USE_ASYNC
// builtin expects: (v4i AS1*, v4i AS3*, imm offset, imm cpol) with GCC vector_size vectors
typedef int v4i_vs __attribute__((vector_size(16)));
typedef __attribute__((address_space(1))) v4i_vs* gptr16_t;
typedef __attribute__((address_space(3))) v4i_vs* lptr16_t;
#endif

// ------------------------------------------------------------------ utilities

__global__ void k_init_uv(float* __restrict__ u, float* __restrict__ v, int n) {
  int i = blockIdx.x * blockDim.x + threadIdx.x;
  if (i < n) { u[i] = 1.0f; v[i] = 1.0f; }
}

__global__ __launch_bounds__(256) void k_rownorm(const float* __restrict__ Z,
                                                 float* __restrict__ out, int cols) {
  __shared__ float red[256];
  const int row = blockIdx.x;
  const float* zr = Z + (size_t)row * cols;
  float s = 0.f;
  for (int j = threadIdx.x * 4; j < cols; j += 256 * 4) {
    float4 a = *(const float4*)(zr + j);
    s += a.x * a.x + a.y * a.y + a.z * a.z + a.w * a.w;
  }
  red[threadIdx.x] = s; __syncthreads();
  for (int st = 128; st > 0; st >>= 1) {
    if (threadIdx.x < st) red[threadIdx.x] += red[threadIdx.x + st];
    __syncthreads();
  }
  if (threadIdx.x == 0) out[row] = red[0];
}

__global__ void k_cvt_bf16(const float* __restrict__ src, __bf16* __restrict__ dst, int n) {
  int i = blockIdx.x * blockDim.x + threadIdx.x;
  if (i < n) dst[i] = (__bf16)src[i];
}

// W1 is (Din=1025, H) row-major; produce W1T (H, DP=1056) bf16, zero-padded in K.
__global__ void k_w1t(const float* __restrict__ W1, __bf16* __restrict__ W1T,
                      int H, int Din, int DP) {
  int i = blockIdx.x * blockDim.x + threadIdx.x;   // over H*DP
  if (i >= H * DP) return;
  int h = i / DP, d = i - h * DP;
  W1T[i] = (d < Din) ? (__bf16)W1[(size_t)d * H + h] : (__bf16)0.f;
}

// W2 is (H, Dout) row-major; produce W2T (Dout, H) bf16.
__global__ void k_w2t(const float* __restrict__ W2, __bf16* __restrict__ W2T,
                      int H, int Dout) {
  int i = blockIdx.x * blockDim.x + threadIdx.x;   // over Dout*H
  if (i >= Dout * H) return;
  int d = i / H, h = i - d * H;
  W2T[i] = (__bf16)W2[(size_t)h * Dout + d];
}

// ------------------------------------------------------------------ Sinkhorn

// y[row] = 1 / (dot(M[row,:], x) + 1e-8). One block per row; M row-major, L2-resident.
__global__ __launch_bounds__(256) void k_rowdot_recip(const float* __restrict__ M,
                                                      const float* __restrict__ x,
                                                      float* __restrict__ y, int n) {
  __shared__ float red[256];
  const int row = blockIdx.x;
  const float* mr = M + (size_t)row * n;
  float s = 0.f;
  for (int j = threadIdx.x * 4; j < n; j += 256 * 4) {
    float4 a = *(const float4*)(mr + j);
    float4 b = *(const float4*)(x + j);
    s += a.x * b.x + a.y * b.y + a.z * b.z + a.w * b.w;
  }
  red[threadIdx.x] = s; __syncthreads();
  for (int st = 128; st > 0; st >>= 1) {
    if (threadIdx.x < st) red[threadIdx.x] += red[threadIdx.x + st];
    __syncthreads();
  }
  if (threadIdx.x == 0) y[row] = 1.0f / (red[0] + 1e-8f);
}

// Per row: argmax_j K[i,j]*v[j] (u_i>0 doesn't affect argmax; ties -> lowest j) and
// ot_part[i] = u[i] * sum_j cost[i,j]*K[i,j]*v[j].
__global__ __launch_bounds__(256) void k_argmax_ot(const float* __restrict__ Kmat,
                                                   const float* __restrict__ cost,
                                                   const float* __restrict__ u,
                                                   const float* __restrict__ v,
                                                   int* __restrict__ idx,
                                                   float* __restrict__ ot_part, int n) {
  __shared__ float rb[256]; __shared__ int ri[256]; __shared__ float rs[256];
  const int row = blockIdx.x, tid = threadIdx.x;
  const float* kr = Kmat + (size_t)row * n;
  const float* cr = cost + (size_t)row * n;
  float best = -1.f; int bi = 0; float s = 0.f;
  for (int j = tid; j < n; j += 256) {
    float p = kr[j] * v[j];
    s += cr[j] * p;
    if (p > best) { best = p; bi = j; }
  }
  rb[tid] = best; ri[tid] = bi; rs[tid] = s; __syncthreads();
  for (int st = 128; st > 0; st >>= 1) {
    if (tid < st) {
      rs[tid] += rs[tid + st];
      float ob = rb[tid + st]; int oi = ri[tid + st];
      if (ob > rb[tid] || (ob == rb[tid] && oi < ri[tid])) { rb[tid] = ob; ri[tid] = oi; }
    }
    __syncthreads();
  }
  if (tid == 0) { idx[row] = ri[0]; ot_part[row] = u[row] * rs[0]; }
}

// Build X = [z_t, t, 0-pad] in bf16 (B x DP) and target = z1[idx] - z0 (f32).
__global__ void k_prep(const float* __restrict__ z0, const float* __restrict__ z1,
                       const float* __restrict__ t, const int* __restrict__ idx,
                       __bf16* __restrict__ X, float* __restrict__ target,
                       int B, int D, int DP) {
  int g = blockIdx.x * blockDim.x + threadIdx.x;   // over B*DP
  if (g >= B * DP) return;
  int i = g / DP, c = g - i * DP;
  if (c < D) {
    int j = idx[i];
    float zm = z1[(size_t)j * D + c];
    float z0v = z0[(size_t)i * D + c];
    float tt = t[i];
    X[g] = (__bf16)((1.0f - tt) * z0v + tt * zm);
    target[(size_t)i * D + c] = zm - z0v;
  } else if (c == D) {
    X[g] = (__bf16)t[i];
  } else {
    X[g] = (__bf16)0.f;
  }
}

// ------------------------------------------------------------------ WMMA GEMM
// C(MxN) = A(MxK) * B(NxK)^T, both bf16 row-major, K%32==0, M%64==0, N%128==0.
// Block: 256 threads = 8 waves in 2(M) x 4(N); block tile 64x128; each wave a 32x32
// output (2x2 accumulators -> 4 independent WMMAs per k-step). LDS double-buffered,
// filled with GLOBAL_LOAD_ASYNC_TO_LDS_B128 (3 chunks/thread/step), ASYNCcnt-pipelined.
// EPI 0: cost/K/KT   EPI 1: relu(+b1) -> bf16 h   EPI 2: (c+b2-target)^2 block partial

template <int EPI>
__global__ __launch_bounds__(256) void k_gemm_bf16(
    const __bf16* __restrict__ A, const __bf16* __restrict__ Bm,
    int lda, int ldb, int kdim,
    const float* __restrict__ e0, const float* __restrict__ e1,
    float* __restrict__ o0, float* __restrict__ o1, float* __restrict__ o2,
    __bf16* __restrict__ obf, int ldo) {
  __shared__ __bf16 As[2][64 * LP];
  __shared__ __bf16 Bs[2][128 * LP];
  __shared__ float  red[256];

  const int tid   = threadIdx.x;
  const int wave  = tid >> 5;
  const int lane  = tid & 31;
  const int waveM = wave >> 2;          // 0..1
  const int waveN = wave & 3;           // 0..3
  const int l16   = lane & 15;
  const int hi    = lane >> 4;

  const int Mbase = blockIdx.y * 64;
  const int Nbase = blockIdx.x * 128;

  // fragment addressing (16-bit A/B layout): lanes 0-15 K in [0..7]&[16..23],
  // lanes 16-31 K in [8..15]&[24..31]
  const int aRow0 = waveM * 32 + l16;
  const int bRow0 = waveN * 32 + l16;
  const int k0    = hi * 8;
  const int k1    = 16 + hi * 8;

  const int S = kdim >> 5;              // k-steps

  // stage one 64x32 A tile + one 128x32 B tile = 768 16B chunks, 3 per thread (uniform)
  auto issue = [&](int step, int buf) {
    const int kk = step * 32;
#pragma unroll
    for (int q = 0; q < 3; ++q) {
      const int c = tid + q * 256;
      const __bf16* g;
      __bf16* l;
      if (c < 256) {                       // A chunks (q==0 for all threads)
        const int row = c >> 2, col = (c & 3) * 8;
        g = A + (size_t)(Mbase + row) * lda + kk + col;
        l = &As[buf][row * LP + col];
      } else {                             // B chunks (q==1,2 for all threads)
        const int cc = c - 256;
        const int row = cc >> 2, col = (cc & 3) * 8;
        g = Bm + (size_t)(Nbase + row) * ldb + kk + col;
        l = &Bs[buf][row * LP + col];
      }
#if USE_ASYNC
      __builtin_amdgcn_global_load_async_to_lds_b128((gptr16_t)g, (lptr16_t)l, 0, 0);
#else
      *(float4*)l = *(const float4*)g;
#endif
    }
  };

  v8f acc[2][2] = {};

  issue(0, 0);
  if (S > 1) issue(1, 1);

  for (int s = 0; s < S; ++s) {
#if USE_ASYNC
    if (s + 1 < S) __builtin_amdgcn_s_wait_asynccnt(3);  // group s done, s+1 may fly
    else           __builtin_amdgcn_s_wait_asynccnt(0);
#endif
    __syncthreads();   // tile (s&1) visible to all waves

    const __bf16* as = As[s & 1];
    const __bf16* bs = Bs[s & 1];
    v16bf a[2], b[2];
    *(float4*)&a[0]           = *(const float4*)&as[aRow0 * LP + k0];
    *(((float4*)&a[0]) + 1)   = *(const float4*)&as[aRow0 * LP + k1];
    *(float4*)&a[1]           = *(const float4*)&as[(aRow0 + 16) * LP + k0];
    *(((float4*)&a[1]) + 1)   = *(const float4*)&as[(aRow0 + 16) * LP + k1];
    *(float4*)&b[0]           = *(const float4*)&bs[bRow0 * LP + k0];
    *(((float4*)&b[0]) + 1)   = *(const float4*)&bs[bRow0 * LP + k1];
    *(float4*)&b[1]           = *(const float4*)&bs[(bRow0 + 16) * LP + k0];
    *(((float4*)&b[1]) + 1)   = *(const float4*)&bs[(bRow0 + 16) * LP + k1];

#pragma unroll
    for (int mi = 0; mi < 2; ++mi)
#pragma unroll
      for (int ni = 0; ni < 2; ++ni)
        acc[mi][ni] = __builtin_amdgcn_wmma_f32_16x16x32_bf16(
            false, a[mi], false, b[ni], (short)0, acc[mi][ni], false, false);

    __syncthreads();   // all waves done reading tile (s&1) before refill
    if (s + 2 < S) issue(s + 2, s & 1);
  }

  float lsum = 0.f;
#pragma unroll
  for (int mi = 0; mi < 2; ++mi) {
#pragma unroll
    for (int ni = 0; ni < 2; ++ni) {
      const int n = Nbase + waveN * 32 + ni * 16 + l16;
#pragma unroll
      for (int r = 0; r < 8; ++r) {
        const int m = Mbase + waveM * 32 + mi * 16 + hi * 8 + r;
        const float c = acc[mi][ni][r];
        if (EPI == 0) {
          float d2  = e0[m] + e1[n] - 2.0f * c;
          float cst = sqrtf(fmaxf(d2, 0.0f));
          float kv  = __expf(cst * -100.0f);     // exp(-cost/0.01)
          o0[(size_t)m * ldo + n] = cst;
          o1[(size_t)m * ldo + n] = kv;
          o2[(size_t)n * ldo + m] = kv;          // K^T for coalesced Sinkhorn v-updates
        } else if (EPI == 1) {
          float h = fmaxf(c + e0[n], 0.0f);
          obf[(size_t)m * ldo + n] = (__bf16)h;
        } else {
          float d = c + e0[n] - e1[(size_t)m * ldo + n];
          lsum += d * d;
        }
      }
    }
  }
  if (EPI == 2) {
    red[tid] = lsum; __syncthreads();
    for (int st = 128; st > 0; st >>= 1) {
      if (tid < st) red[tid] += red[tid + st];
      __syncthreads();
    }
    if (tid == 0) o0[blockIdx.y * gridDim.x + blockIdx.x] = red[0];
  }
}

// ------------------------------------------------------------------ finalize

__global__ __launch_bounds__(256) void k_finalize(const float* __restrict__ lp, int nl,
                                                  const float* __restrict__ op, int no,
                                                  float* __restrict__ out, float scale) {
  __shared__ float red[256];
  const int tid = threadIdx.x;
  float s = 0.f;
  for (int j = tid; j < nl; j += 256) s += lp[j];
  red[tid] = s; __syncthreads();
  for (int st = 128; st > 0; st >>= 1) { if (tid < st) red[tid] += red[tid + st]; __syncthreads(); }
  if (tid == 0) out[0] = red[0] * scale;
  __syncthreads();
  s = 0.f;
  for (int j = tid; j < no; j += 256) s += op[j];
  red[tid] = s; __syncthreads();
  for (int st = 128; st > 0; st >>= 1) { if (tid < st) red[tid] += red[tid + st]; __syncthreads(); }
  if (tid == 0) out[1] = red[0];
}

// ------------------------------------------------------------------ host

extern "C" void kernel_launch(void* const* d_in, const int* in_sizes, int n_in,
                              void* d_out, int out_size, void* d_ws, size_t ws_size,
                              hipStream_t stream) {
  constexpr int B_ = 2048, D_ = 1024, H_ = 4096, DP_ = 1056;  // DP_ = D_+1 padded to /32

  const float* z0 = (const float*)d_in[0];
  const float* z1 = (const float*)d_in[1];
  const float* tv = (const float*)d_in[2];
  const float* W1 = (const float*)d_in[3];
  const float* b1 = (const float*)d_in[4];
  const float* W2 = (const float*)d_in[5];
  const float* b2 = (const float*)d_in[6];
  float* out = (float*)d_out;

  size_t off = 0;
  auto alloc = [&](size_t bytes) -> void* {
    void* p = (char*)d_ws + off;
    off = (off + bytes + 255) & ~(size_t)255;
    return p;
  };
  float*  costM     = (float*) alloc((size_t)B_ * B_ * 4);
  float*  Kmat      = (float*) alloc((size_t)B_ * B_ * 4);
  float*  KT        = (float*) alloc((size_t)B_ * B_ * 4);
  float*  n0        = (float*) alloc((size_t)B_ * 4);
  float*  n1        = (float*) alloc((size_t)B_ * 4);
  float*  u         = (float*) alloc((size_t)B_ * 4);
  float*  v         = (float*) alloc((size_t)B_ * 4);
  int*    idx       = (int*)   alloc((size_t)B_ * 4);
  float*  target    = (float*) alloc((size_t)B_ * D_ * 4);
  __bf16* z0b       = (__bf16*)alloc((size_t)B_ * D_ * 2);
  __bf16* z1b       = (__bf16*)alloc((size_t)B_ * D_ * 2);
  __bf16* Xb        = (__bf16*)alloc((size_t)B_ * DP_ * 2);
  __bf16* W1T       = (__bf16*)alloc((size_t)H_ * DP_ * 2);
  __bf16* W2T       = (__bf16*)alloc((size_t)D_ * H_ * 2);
  __bf16* hb        = (__bf16*)alloc((size_t)B_ * H_ * 2);
  float*  loss_part = (float*) alloc((size_t)(D_ / 128) * (B_ / 64) * 4);
  float*  ot_part   = (float*) alloc((size_t)B_ * 4);

  // prep: norms, conversions, transposes, u=v=1
  k_init_uv<<<(B_ + 255) / 256, 256, 0, stream>>>(u, v, B_);
  k_rownorm<<<B_, 256, 0, stream>>>(z0, n0, D_);
  k_rownorm<<<B_, 256, 0, stream>>>(z1, n1, D_);
  k_cvt_bf16<<<(B_ * D_) / 256, 256, 0, stream>>>(z0, z0b, B_ * D_);
  k_cvt_bf16<<<(B_ * D_) / 256, 256, 0, stream>>>(z1, z1b, B_ * D_);
  k_w1t<<<(H_ * DP_) / 256, 256, 0, stream>>>(W1, W1T, H_, D_ + 1, DP_);
  k_w2t<<<(D_ * H_) / 256, 256, 0, stream>>>(W2, W2T, H_, D_);

  // cost matrix: G = z0*z1^T via WMMA; epilogue writes cost, K, K^T
  dim3 gC(B_ / 128, B_ / 64);
  k_gemm_bf16<0><<<gC, 256, 0, stream>>>(z0b, z1b, D_, D_, D_,
                                         n0, n1, costM, Kmat, KT, nullptr, B_);

  // Sinkhorn: 200 L2-resident matvecs (K and K^T both row-major, 32MB << 192MB L2)
  for (int it = 0; it < SINK_ITERS; ++it) {
    k_rowdot_recip<<<B_, 256, 0, stream>>>(Kmat, v, u, B_);
    k_rowdot_recip<<<B_, 256, 0, stream>>>(KT,  u, v, B_);
  }

  // transport-plan argmax + ot_cost partials; then build MLP input + target
  k_argmax_ot<<<B_, 256, 0, stream>>>(Kmat, costM, u, v, idx, ot_part, B_);
  k_prep<<<(B_ * DP_) / 256, 256, 0, stream>>>(z0, z1, tv, idx, Xb, target, B_, D_, DP_);

  // MLP layer 1: h = relu(X*W1 + b1)  -> bf16
  dim3 g1(H_ / 128, B_ / 64);
  k_gemm_bf16<1><<<g1, 256, 0, stream>>>(Xb, W1T, DP_, DP_, DP_,
                                         b1, nullptr, nullptr, nullptr, nullptr, hb, H_);

  // MLP layer 2 + fused loss partials: sum (h*W2 + b2 - target)^2 per block
  dim3 g2(D_ / 128, B_ / 64);
  k_gemm_bf16<2><<<g2, 256, 0, stream>>>(hb, W2T, H_, H_, H_,
                                         b2, target, loss_part, nullptr, nullptr, nullptr, D_);

  // deterministic final reduction -> (loss, ot_cost)
  k_finalize<<<1, 256, 0, stream>>>(loss_part, (D_ / 128) * (B_ / 64),
                                    ot_part, B_, out, 1.0f / ((float)B_ * (float)D_));

  (void)in_sizes; (void)n_in; (void)out_size; (void)ws_size;
}
